// SplitBrainAttention_46059229282564
// MI455X (gfx1250) — matrix operand order repaired and checked by reading, not compiled, not run on hardware
//
#include <hip/hip_runtime.h>
#include <hip/hip_bf16.h>

// ---------------- problem constants ----------------
constexpr int Bb = 2, Tt = 2048, Dd = 2048, Nn = 16, KH = 8, Hh = 128;
constexpr int GQ = Nn / KH;           // 2
constexpr int WIN = 1024;
constexpr float SOFT_CAP = 50.0f;
constexpr float Q_SCALE_F = 0.08838834764831845f;   // 128^-0.5

typedef __attribute__((ext_vector_type(16))) __bf16 v16bf;
typedef __attribute__((ext_vector_type(8)))  float  v8f;

union Frag { v16bf v; uint4 u[2]; };

__device__ __forceinline__ v8f wmma_bf16(v16bf a, v16bf b, v8f c) {
    return __builtin_amdgcn_wmma_f32_16x16x32_bf16(
        false, a, false, b, (short)0, c, false, false);
}

__device__ __forceinline__ float softcap50(float s) {
    float x  = s * (1.0f / SOFT_CAP);
    float ax = fminf(fabsf(x), 15.0f);
    float e  = __expf(2.0f * ax);
    float t  = (e - 1.0f) / (e + 1.0f);
    return SOFT_CAP * copysignf(t, x);
}

// async copy of 16 bytes global -> LDS (per-lane addresses, ASYNCcnt-tracked)
__device__ __forceinline__ void async_g2l_b128(void* lds, const void* gaddr) {
    unsigned l = (unsigned)(unsigned long long)lds;   // low 32 bits = LDS offset
    asm volatile("global_load_async_to_lds_b128 %0, %1, off"
                 :: "v"(l), "v"(gaddr) : "memory");
}
__device__ __forceinline__ void wait_async0() {
    asm volatile("s_wait_asynccnt 0x0" ::: "memory");
}

// ---------------- elementwise converts ----------------
__global__ void sba_cvt_f32_bf16(const float* __restrict__ s,
                                 __bf16* __restrict__ d, long n) {
    long i = (long)blockIdx.x * blockDim.x + threadIdx.x;
    long st = (long)gridDim.x * blockDim.x;
    for (; i < n; i += st) d[i] = (__bf16)s[i];
}

// out[z][c][r] = in[z][r][c]  (weights [*, rows, cols] -> bf16 [*, cols, rows])
__global__ void sba_transpose_bf16(const float* __restrict__ s,
                                   __bf16* __restrict__ d,
                                   int rows, int cols, long n) {
    long i = (long)blockIdx.x * blockDim.x + threadIdx.x;
    long st = (long)gridDim.x * blockDim.x;
    long mat = (long)rows * cols;
    for (; i < n; i += st) {
        long z = i / mat, rem = i % mat;
        int r = (int)(rem / cols), c = (int)(rem % cols);
        d[z * mat + (long)c * rows + r] = (__bf16)s[i];
    }
}

// ---------------- generic bf16 WMMA GEMM (pipelined, 4-wide B groups) ----------------
// C[M, Nc] (f32) = A[M, Kd] (bf16, row-major lda) * WT[Nc, Kd]^T (bf16, row-major ldwt)
// grid: (M/128, Nc/128, nz*nhead); block: 256 (8 waves, 16 rows each). Kd % 64 == 0.
__global__ __launch_bounds__(256, 1)
void sba_gemm_bf16(const __bf16* __restrict__ A,
                   const __bf16* __restrict__ WT,
                   float* __restrict__ C,
                   int Kd, int lda, int ldwt, int ldc,
                   int nhead, long As_b, long Ws_h,
                   long Cs_b, long Cs_h) {
    int z = blockIdx.z;
    int bz = z / nhead, h = z % nhead;
    A  += (long)bz * As_b;
    WT += (long)h  * Ws_h;
    C  += (long)bz * Cs_b + (long)h * Cs_h;

    int wave = threadIdx.x >> 5;
    int lane = threadIdx.x & 31;
    int nlo  = lane & 15, hi = lane >> 4;
    int m0 = blockIdx.x * 128 + wave * 16;
    int c0 = blockIdx.y * 128;

    v8f acc[8] = {};
    // A layout: lanes 0-15 hold K {0..7},{16..23}; lanes 16-31 {8..15},{24..31}
    const __bf16* arow = A  + (long)(m0 + nlo) * lda  + hi * 8;
    // B layout: lane (&15) = column, K {0..7},{16..23} for lanes 0-15; {8..15},{24..31} for 16-31
    const __bf16* wrow = WT + (long)(c0 + nlo) * ldwt + hi * 8;
    const long wstep = (long)16 * ldwt;

    Frag afA, afB, bg0[4], bg1[4];
    auto loadAf = [&](Frag& af, int k0) {
        af.u[0] = *(const uint4*)(arow + k0);
        af.u[1] = *(const uint4*)(arow + k0 + 16);
    };
    auto loadB4 = [&](Frag* bf, int k0, int ctbase) {
        #pragma unroll
        for (int i = 0; i < 4; ++i) {
            const __bf16* w = wrow + (ctbase + i) * wstep + k0;
            bf[i].u[0] = *(const uint4*)(w);
            bf[i].u[1] = *(const uint4*)(w + 16);
        }
    };

    loadAf(afA, 0);
    loadB4(bg0, 0, 0);
    for (int k0 = 0; k0 < Kd; k0 += 64) {
        loadB4(bg1, k0, 4);
        #pragma unroll
        for (int i = 0; i < 4; ++i) acc[i] = wmma_bf16(afA.v, bg0[i].v, acc[i]);
        loadAf(afB, k0 + 32);
        loadB4(bg0, k0 + 32, 0);
        #pragma unroll
        for (int i = 0; i < 4; ++i) acc[4 + i] = wmma_bf16(afA.v, bg1[i].v, acc[4 + i]);
        loadB4(bg1, k0 + 32, 4);
        #pragma unroll
        for (int i = 0; i < 4; ++i) acc[i] = wmma_bf16(afB.v, bg0[i].v, acc[i]);
        if (k0 + 64 < Kd) { loadAf(afA, k0 + 64); loadB4(bg0, k0 + 64, 0); }
        #pragma unroll
        for (int i = 0; i < 4; ++i) acc[4 + i] = wmma_bf16(afB.v, bg1[i].v, acc[4 + i]);
    }

    #pragma unroll
    for (int ct = 0; ct < 8; ++ct)
        #pragma unroll
        for (int r = 0; r < 8; ++r)
            C[(long)(m0 + r + hi * 8) * ldc + c0 + ct * 16 + nlo] = acc[ct][r];
}

// ---------------- RMSNorm + RoPE (+scale) : f32 -> bf16 ----------------
__global__ void sba_normrope(const float* __restrict__ src,
                             __bf16* __restrict__ dst,
                             const float* __restrict__ nscale,
                             const int* __restrict__ segpos,
                             int heads, float outscale) {
    int wid  = blockIdx.x * (blockDim.x >> 5) + (threadIdx.x >> 5);
    int lane = threadIdx.x & 31;
    int t = (wid / heads) % Tt;
    int b = wid / (heads * Tt);

    const float* row = src + (long)wid * Hh;
    __bf16* out = dst + (long)wid * Hh;

    int i = lane * 2;
    float x0 = row[i], x1 = row[i + 1];
    float y0 = row[i + 64], y1 = row[i + 65];
    float ss = x0 * x0 + x1 * x1 + y0 * y0 + y1 * y1;
    #pragma unroll
    for (int mask = 1; mask <= 16; mask <<= 1) ss += __shfl_xor(ss, mask, 32);
    float rstd = rsqrtf(ss * (1.0f / 128.0f) + 1e-6f);

    float n0 = x0 * rstd * (1.0f + nscale[i]);
    float n1 = x1 * rstd * (1.0f + nscale[i + 1]);
    float n2 = y0 * rstd * (1.0f + nscale[i + 64]);
    float n3 = y1 * rstd * (1.0f + nscale[i + 65]);

    float pos = (float)segpos[b * Tt + t];
    float a0 = pos * __powf(10000.0f, -(float)i * (1.0f / 64.0f));
    float a1 = pos * __powf(10000.0f, -(float)(i + 1) * (1.0f / 64.0f));
    float c0 = __cosf(a0), s0 = __sinf(a0);
    float c1 = __cosf(a1), s1 = __sinf(a1);

    out[i]      = (__bf16)((n0 * c0 - n2 * s0) * outscale);
    out[i + 1]  = (__bf16)((n1 * c1 - n3 * s1) * outscale);
    out[i + 64] = (__bf16)((n2 * c0 + n0 * s0) * outscale);
    out[i + 65] = (__bf16)((n3 * c1 + n1 * s1) * outscale);
}

// ---------------- flash attention, sliding window, soft-cap ----------------
// grid (T/32, KH, B); block 128 (4 waves). waves {0,1} -> q-head g0, {2,3} -> g1.
__global__ __launch_bounds__(128, 1)
void sba_attn(const __bf16* __restrict__ Qb,
              const __bf16* __restrict__ Kb,
              const __bf16* __restrict__ Vb,
              __bf16* __restrict__ Enc) {
    __shared__ __align__(16) __bf16 Kt[32][128];        // K tile, natural [key][h] (async-staged)
    __shared__ __align__(16) __bf16 Vt[128][40];        // V tile transposed [h][key]
    __shared__ __align__(16) __bf16 Pl[4][16][40];      // per-wave P staging

    int q0 = blockIdx.x * 32;
    int kh = blockIdx.y;
    int b  = blockIdx.z;
    int wave = threadIdx.x >> 5;
    int lane = threadIdx.x & 31;
    int n    = kh * GQ + (wave >> 1);
    int qrow0 = q0 + (wave & 1) * 16;
    int nlo = lane & 15, hi = lane >> 4;

    // Q fragments: 16 rows x 128 (4 A-frags)
    Frag qf[4];
    const __bf16* qrow = Qb + (((long)(b * Tt + qrow0 + nlo) * Nn) + n) * Hh;
    #pragma unroll
    for (int c = 0; c < 4; ++c) {
        qf[c].u[0] = *(const uint4*)(qrow + c * 32 + hi * 8);
        qf[c].u[1] = *(const uint4*)(qrow + c * 32 + 16 + hi * 8);
    }

    v8f O[8] = {};
    float m[8], l[8];
    #pragma unroll
    for (int r = 0; r < 8; ++r) { m[r] = -1e30f; l[r] = 0.0f; }

    int kstart = q0 - (WIN - 1); if (kstart < 0) kstart = 0; kstart &= ~31;
    int kend = q0 + 31;

    for (int kb = kstart; kb <= kend; kb += 32) {
        // ---- async-stage K tile (32 keys x 128 h, natural layout) ----
        #pragma unroll
        for (int it = 0; it < 4; ++it) {
            int c = threadIdx.x + it * 128;               // 512 x 16B chunks
            int key = c >> 4, hseg = (c & 15) * 8;
            const __bf16* g = Kb + (((long)(b * Tt + kb + key) * KH) + kh) * Hh + hseg;
            async_g2l_b128(&Kt[key][hseg], g);
        }
        // ---- stage V tile transposed: Vt[h][key] (VALU path, needs transpose) ----
        for (int i = threadIdx.x; i < 32 * 16; i += blockDim.x) {
            int key = i >> 4, hseg = (i & 15) * 8;
            const __bf16* vp = Vb + (((long)(b * Tt + kb + key) * KH) + kh) * Hh + hseg;
            uint4 pk = *(const uint4*)vp;
            const __bf16* e = (const __bf16*)&pk;
            #pragma unroll
            for (int j = 0; j < 8; ++j) Vt[hseg + j][key] = e[j];
        }
        wait_async0();
        __syncthreads();

        // ---- S = Q * K^T (two 16x16 tiles over 32 keys) ----
        const __bf16* krow0 = &Kt[nlo][hi * 8];
        const __bf16* krow1 = &Kt[16 + nlo][hi * 8];
        v8f s0 = {}, s1 = {};
        #pragma unroll
        for (int g = 0; g < 2; ++g) {
            Frag kf0[2], kf1[2];
            #pragma unroll
            for (int i = 0; i < 2; ++i) {
                int c = g * 2 + i;
                kf0[i].u[0] = *(const uint4*)(krow0 + c * 32);
                kf0[i].u[1] = *(const uint4*)(krow0 + c * 32 + 16);
                kf1[i].u[0] = *(const uint4*)(krow1 + c * 32);
                kf1[i].u[1] = *(const uint4*)(krow1 + c * 32 + 16);
            }
            #pragma unroll
            for (int i = 0; i < 2; ++i) {
                int c = g * 2 + i;
                s0 = wmma_bf16(qf[c].v, kf0[i].v, s0);
                s1 = wmma_bf16(qf[c].v, kf1[i].v, s1);
            }
        }

        // ---- soft-cap + window/causal mask + online softmax ----
        float p0[8], p1[8], rm[8];
        #pragma unroll
        for (int r = 0; r < 8; ++r) {
            int qp  = qrow0 + r + hi * 8;
            int kp0 = kb + nlo, kp1 = kb + 16 + nlo;
            float v0 = softcap50(s0[r]);
            float v1 = softcap50(s1[r]);
            p0[r] = ((kp0 <= qp) && (kp0 > qp - WIN)) ? v0 : -1e30f;
            p1[r] = ((kp1 <= qp) && (kp1 > qp - WIN)) ? v1 : -1e30f;
            rm[r] = fmaxf(p0[r], p1[r]);
        }
        #pragma unroll
        for (int mask = 1; mask <= 8; mask <<= 1)
            #pragma unroll
            for (int r = 0; r < 8; ++r)
                rm[r] = fmaxf(rm[r], __shfl_xor(rm[r], mask, 32));

        float sc[8], rs[8];
        #pragma unroll
        for (int r = 0; r < 8; ++r) {
            float mn = fmaxf(m[r], rm[r]);
            sc[r] = __expf(m[r] - mn);
            m[r]  = mn;
            p0[r] = __expf(p0[r] - mn);
            p1[r] = __expf(p1[r] - mn);
            rs[r] = p0[r] + p1[r];
        }
        #pragma unroll
        for (int mask = 1; mask <= 8; mask <<= 1)
            #pragma unroll
            for (int r = 0; r < 8; ++r)
                rs[r] += __shfl_xor(rs[r], mask, 32);
        #pragma unroll
        for (int r = 0; r < 8; ++r) l[r] = l[r] * sc[r] + rs[r];
        #pragma unroll
        for (int ct = 0; ct < 8; ++ct)
            #pragma unroll
            for (int r = 0; r < 8; ++r) O[ct][r] *= sc[r];

        // ---- P: C-layout -> A-layout via per-wave LDS bounce ----
        #pragma unroll
        for (int r = 0; r < 8; ++r) {
            Pl[wave][r + hi * 8][nlo]      = (__bf16)p0[r];
            Pl[wave][r + hi * 8][nlo + 16] = (__bf16)p1[r];
        }
        Frag pf;
        pf.u[0] = *(const uint4*)&Pl[wave][nlo][hi * 8];
        pf.u[1] = *(const uint4*)&Pl[wave][nlo][16 + hi * 8];

        #pragma unroll
        for (int g = 0; g < 2; ++g) {
            Frag vfr[4];
            #pragma unroll
            for (int i = 0; i < 4; ++i) {
                int ct = g * 4 + i;
                vfr[i].u[0] = *(const uint4*)&Vt[ct * 16 + nlo][hi * 8];
                vfr[i].u[1] = *(const uint4*)&Vt[ct * 16 + nlo][16 + hi * 8];
            }
            #pragma unroll
            for (int i = 0; i < 4; ++i)
                O[g * 4 + i] = wmma_bf16(pf.v, vfr[i].v, O[g * 4 + i]);
        }
        __syncthreads();
    }

    #pragma unroll
    for (int r = 0; r < 8; ++r) l[r] = 1.0f / l[r];
    #pragma unroll
    for (int ct = 0; ct < 8; ++ct)
        #pragma unroll
        for (int r = 0; r < 8; ++r) {
            long row = (long)(b * Tt + qrow0 + r + hi * 8);
            Enc[(row * Nn + n) * Hh + ct * 16 + nlo] = (__bf16)(O[ct][r] * l[r]);
        }
}

// ---------------- host orchestration ----------------
extern "C" void kernel_launch(void* const* d_in, const int* in_sizes, int n_in,
                              void* d_out, int out_size, void* d_ws, size_t ws_size,
                              hipStream_t stream) {
    const float* x    = (const float*)d_in[0];
    const float* w_q  = (const float*)d_in[1];
    const float* w_kv = (const float*)d_in[2];
    const float* w_o  = (const float*)d_in[3];
    const float* qns  = (const float*)d_in[4];
    const float* kns  = (const float*)d_in[5];
    const int*   spos = (const int*)d_in[6];
    float* out = (float*)d_out;
    (void)in_sizes; (void)n_in; (void)out_size; (void)ws_size;

    char* cur = (char*)d_ws;
    auto carve = [&](long bytes) {
        void* p = cur;
        cur += (bytes + 255) & ~255L;
        return p;
    };
    const long XB_E  = (long)Bb * Tt * Dd;
    const long WQT_E = (long)Nn * Hh * Dd;
    const long WKT_E = (long)KH * Hh * Dd;
    const long WOT_E = (long)Dd * Nn * Hh;
    const long QF_E  = (long)Bb * Tt * Nn * Hh;
    const long KF_E  = (long)Bb * Tt * KH * Hh;

    __bf16* xb  = (__bf16*)carve(XB_E * 2);
    __bf16* wqT = (__bf16*)carve(WQT_E * 2);
    __bf16* wkT = (__bf16*)carve(WKT_E * 2);
    __bf16* wvT = (__bf16*)carve(WKT_E * 2);
    __bf16* woT = (__bf16*)carve(WOT_E * 2);
    float*  qf  = (float*)carve(QF_E * 4);
    float*  kf  = (float*)carve(KF_E * 4);
    float*  vf  = (float*)carve(KF_E * 4);
    __bf16* qb  = (__bf16*)carve(QF_E * 2);
    __bf16* kb  = (__bf16*)carve(KF_E * 2);
    __bf16* vb  = (__bf16*)carve(KF_E * 2);
    __bf16* enc = (__bf16*)carve(QF_E * 2);

    sba_cvt_f32_bf16<<<4096, 256, 0, stream>>>(x, xb, XB_E);
    sba_transpose_bf16<<<4096, 256, 0, stream>>>(w_q, wqT, Dd, Hh, WQT_E);
    sba_transpose_bf16<<<2048, 256, 0, stream>>>(w_kv, wkT, Dd, Hh, WKT_E);
    sba_transpose_bf16<<<2048, 256, 0, stream>>>(w_kv + (long)KH * Dd * Hh, wvT, Dd, Hh, WKT_E);
    sba_transpose_bf16<<<4096, 256, 0, stream>>>(w_o, woT, Nn * Hh, Dd, WOT_E);

    sba_gemm_bf16<<<dim3(Tt / 128, 1, Bb * Nn), 256, 0, stream>>>(
        xb, wqT, qf, Dd, Dd, Dd, Nn * Hh, Nn,
        (long)Tt * Dd, (long)Hh * Dd, (long)Tt * Nn * Hh, (long)Hh);
    sba_gemm_bf16<<<dim3(Tt / 128, 1, Bb * KH), 256, 0, stream>>>(
        xb, wkT, kf, Dd, Dd, Dd, KH * Hh, KH,
        (long)Tt * Dd, (long)Hh * Dd, (long)Tt * KH * Hh, (long)Hh);
    sba_gemm_bf16<<<dim3(Tt / 128, 1, Bb * KH), 256, 0, stream>>>(
        xb, wvT, vf, Dd, Dd, Dd, KH * Hh, KH,
        (long)Tt * Dd, (long)Hh * Dd, (long)Tt * KH * Hh, (long)Hh);

    sba_normrope<<<(Bb * Tt * Nn) / 8, 256, 0, stream>>>(qf, qb, qns, spos, Nn, Q_SCALE_F);
    sba_normrope<<<(Bb * Tt * KH) / 8, 256, 0, stream>>>(kf, kb, kns, spos, KH, 1.0f);
    sba_cvt_f32_bf16<<<2048, 256, 0, stream>>>(vf, vb, KF_E);

    sba_attn<<<dim3(Tt / 32, KH, Bb), 128, 0, stream>>>(qb, kb, vb, enc);

    sba_gemm_bf16<<<dim3((Bb * Tt) / 128, Dd / 128, 1), 256, 0, stream>>>(
        enc, woT, out, Nn * Hh, Nn * Hh, Nn * Hh, Dd, 1,
        0L, 0L, 0L, 0L);
}
